// MPSelfAttention_8942121910753
// MI455X (gfx1250) — matrix operand order repaired and checked by reading, not compiled, not run on hardware
//
#include <hip/hip_runtime.h>
#include <hip/hip_bf16.h>

// ---------------------------------------------------------------------------
// Types for CDNA5 WMMA (wave32)
// ---------------------------------------------------------------------------
typedef __bf16 bf16_t;
typedef __attribute__((ext_vector_type(16))) __bf16 v16bf;
typedef __attribute__((ext_vector_type(8)))  __bf16 v8bf;
typedef __attribute__((ext_vector_type(8)))  float  v8f;

#define EPS 1e-6f

__device__ inline bf16_t f2b(float f) {
    unsigned u = __builtin_bit_cast(unsigned, f);
    unsigned r = (u + 0x7FFFu + ((u >> 16) & 1u)) >> 16;   // round-to-nearest-even
    return __builtin_bit_cast(bf16_t, (unsigned short)r);
}
__device__ inline float b2f(bf16_t b) {
    unsigned short s = __builtin_bit_cast(unsigned short, b);
    return __builtin_bit_cast(float, (unsigned)s << 16);
}
__device__ inline v8f zero8() {
    v8f z = {0.f,0.f,0.f,0.f,0.f,0.f,0.f,0.f};
    return z;
}

// ---------------------------------------------------------------------------
// CDNA5 async global->LDS copy (GLOBAL_LOAD_ASYNC_TO_LDS_B128, ASYNCcnt).
// The LDS destination address is the low 32 bits of the generic pointer
// (flat-LDS aperture, wave-relative).
// ---------------------------------------------------------------------------
__device__ inline void async_copy16(const bf16_t* g, bf16_t* l) {
    const unsigned lds = (unsigned)(unsigned long long)l;
    asm volatile("global_load_async_to_lds_b128 %0, %1, off"
                 :: "v"(lds), "v"((unsigned long long)g)
                 : "memory");
}
__device__ inline void wait_async_0() { asm volatile("s_wait_asynccnt 0x0" ::: "memory"); }
__device__ inline void wait_async_4() { asm volatile("s_wait_asynccnt 0x4" ::: "memory"); }

// A fragment, 16x32 bf16, row-major tile in LDS, ld in elements.
// ISA 7.12.2: lanes 0-15 -> M=lane, K in {0..7,16..23}; lanes 16-31 -> K in {8..15,24..31}
__device__ inline v16bf frag_A(const bf16_t* tile, int ld, int k0, int lane) {
    int half = lane >> 4;
    int m    = lane & 15;
    const bf16_t* p = tile + m * ld + k0 + half * 8;
    v8bf lo = *(const v8bf*)p;
    v8bf hi = *(const v8bf*)(p + 16);
    v16bf r;
#pragma unroll
    for (int i = 0; i < 8; ++i) { r[i] = lo[i]; r[i + 8] = hi[i]; }
    return r;
}

// B fragment, 32x16 bf16 stored as W[n][k] row-major (ld in elements).
// lanes 0-15 -> N=lane, K in {0..7,16..23}; lanes 16-31 -> N=lane-16, K in {8..15,24..31}
__device__ inline v16bf frag_B(const bf16_t* tile, int ld, int k0, int lane) {
    int half = lane >> 4;
    int n    = lane & 15;
    const bf16_t* p = tile + n * ld + k0 + half * 8;
    v8bf lo = *(const v8bf*)p;
    v8bf hi = *(const v8bf*)(p + 16);
    v16bf r;
#pragma unroll
    for (int i = 0; i < 8; ++i) { r[i] = lo[i]; r[i + 8] = hi[i]; }
    return r;
}

__device__ inline v8f wmma_bf16(v16bf a, v16bf b, v8f c) {
    return __builtin_amdgcn_wmma_f32_16x16x32_bf16(
        /*neg_a=*/false, a, /*neg_b=*/false, b,
        /*c_mod=*/(short)0, c, /*reuse_a=*/false, /*reuse_b=*/false);
}

// ---------------------------------------------------------------------------
// Kernel 1: unit_norm rows of W [rows x K] -> bf16
// ---------------------------------------------------------------------------
__global__ void __launch_bounds__(256) normw_kernel(const float* __restrict__ w,
                                                    bf16_t* __restrict__ out, int K) {
    __shared__ float red[256];
    const int row = blockIdx.x;
    const float* r = w + (size_t)row * K;
    float s = 0.f;
    for (int i = threadIdx.x; i < K; i += 256) { float v = r[i]; s += v * v; }
    red[threadIdx.x] = s;
    __syncthreads();
    for (int off = 128; off > 0; off >>= 1) {
        if ((int)threadIdx.x < off) red[threadIdx.x] += red[threadIdx.x + off];
        __syncthreads();
    }
    const float scale = rsqrtf(red[0] + EPS);   // unit_norm: sum (not mean)
    bf16_t* o = out + (size_t)row * K;
    for (int i = threadIdx.x; i < K; i += 256) o[i] = f2b(r[i] * scale);
}

// ---------------------------------------------------------------------------
// Kernel 2: f32 -> bf16 elementwise
// ---------------------------------------------------------------------------
__global__ void __launch_bounds__(256) cvt_kernel(const float* __restrict__ in,
                                                  bf16_t* __restrict__ out, long n) {
    long i = (long)blockIdx.x * 256 + threadIdx.x;
    if (i < n) out[i] = f2b(in[i]);
}

// ---------------------------------------------------------------------------
// Kernel 3: bf16 WMMA GEMM  C[M,N] = A[M,K] * W[N,K]^T
// workgroup tile 128x128, 8 waves (4x2), wave tile 32x64, K-step 32.
// Double-buffered LDS filled with async global->LDS copies (ASYNCcnt
// pipelining: issue slab k+1, s_wait_asynccnt 4 retires slab k).
// ---------------------------------------------------------------------------
template <bool OUT_BF16>
__global__ void __launch_bounds__(256) gemm_wmma(const bf16_t* __restrict__ A,
                                                 const bf16_t* __restrict__ W,
                                                 float* __restrict__ Cf,
                                                 bf16_t* __restrict__ Cb,
                                                 int M, int N, int K) {
    __shared__ bf16_t As[2][128 * 32];
    __shared__ bf16_t Ws[2][128 * 32];
    const int tid  = threadIdx.x;
    const int lane = tid & 31;
    const int wave = tid >> 5;       // 0..7
    const int wm   = wave >> 1;      // 0..3  -> 32 rows each
    const int wn   = wave & 1;       // 0..1  -> 64 cols each
    const int nbx  = N >> 7;
    const int m0   = (blockIdx.x / nbx) << 7;
    const int n0   = (blockIdx.x % nbx) << 7;

    v8f acc[2][4];
#pragma unroll
    for (int i = 0; i < 2; ++i)
#pragma unroll
        for (int j = 0; j < 4; ++j) acc[i][j] = zero8();

    const int ldr = tid >> 2;          // 0..63
    const int ldc = (tid & 3) * 8;     // 0,8,16,24

    auto issue = [&](int buf, int k0) {
        async_copy16(&A[(size_t)(m0 + ldr) * K + k0 + ldc],      &As[buf][ldr * 32 + ldc]);
        async_copy16(&A[(size_t)(m0 + ldr + 64) * K + k0 + ldc], &As[buf][(ldr + 64) * 32 + ldc]);
        async_copy16(&W[(size_t)(n0 + ldr) * K + k0 + ldc],      &Ws[buf][ldr * 32 + ldc]);
        async_copy16(&W[(size_t)(n0 + ldr + 64) * K + k0 + ldc], &Ws[buf][(ldr + 64) * 32 + ldc]);
    };

    issue(0, 0);
    for (int k0 = 0; k0 < K; k0 += 32) {
        const int buf = (k0 >> 5) & 1;
        if (k0 + 32 < K) {
            issue(buf ^ 1, k0 + 32);   // prefetch next slab into other buffer
            wait_async_4();            // retire current slab (4 newest still in flight)
        } else {
            wait_async_0();
        }
        __syncthreads();

        v16bf af0 = frag_A(&As[buf][(wm * 32) * 32], 32, 0, lane);
        v16bf af1 = frag_A(&As[buf][(wm * 32 + 16) * 32], 32, 0, lane);
#pragma unroll
        for (int nt = 0; nt < 4; ++nt) {
            v16bf bw = frag_B(&Ws[buf][(wn * 64 + nt * 16) * 32], 32, 0, lane);
            acc[0][nt] = wmma_bf16(af0, bw, acc[0][nt]);
            acc[1][nt] = wmma_bf16(af1, bw, acc[1][nt]);
        }
        __syncthreads();
    }

    const int cr = (lane < 16) ? 0 : 8;
    const int cc = lane & 15;
#pragma unroll
    for (int mt = 0; mt < 2; ++mt)
#pragma unroll
        for (int nt = 0; nt < 4; ++nt) {
            const int row = m0 + wm * 32 + mt * 16 + cr;
            const int col = n0 + wn * 64 + nt * 16 + cc;
#pragma unroll
            for (int g = 0; g < 8; ++g) {
                float v = acc[mt][nt][g];
                if (OUT_BF16) Cb[(size_t)(row + g) * N + col] = f2b(v);
                else          Cf[(size_t)(row + g) * N + col] = v;
            }
        }
}

// ---------------------------------------------------------------------------
// Kernel 4: split qkv [B,S,3,H,hd] bf16, rms_norm q/k over hd=128,
// fold 1/sqrt(hd) into q, emit q,k [B,H,S,128] bf16 and vT [B,H,128,S] bf16.
// One block (128 threads) per (b,h,s).
// ---------------------------------------------------------------------------
__global__ void __launch_bounds__(128) qkv_prep(const bf16_t* __restrict__ qkv,
                                                bf16_t* __restrict__ q,
                                                bf16_t* __restrict__ k,
                                                bf16_t* __restrict__ vT) {
    __shared__ float red[128];
    const int d   = threadIdx.x;
    const int idx = blockIdx.x;
    const int s   = idx & 2047;     // S = 2048
    const int bh  = idx >> 11;      // b*16 + h
    const int h   = bh & 15;
    const int b   = bh >> 4;

    const size_t rowbase = ((size_t)(b * 2048 + s)) * 6144;
    const float qv = b2f(qkv[rowbase + 0 * 2048 + h * 128 + d]);
    const float kv = b2f(qkv[rowbase + 1 * 2048 + h * 128 + d]);
    const float vv = b2f(qkv[rowbase + 2 * 2048 + h * 128 + d]);

    red[d] = qv * qv; __syncthreads();
    for (int off = 64; off > 0; off >>= 1) { if (d < off) red[d] += red[d + off]; __syncthreads(); }
    const float rq = rsqrtf(red[0] * (1.f / 128.f) + EPS);
    __syncthreads();
    red[d] = kv * kv; __syncthreads();
    for (int off = 64; off > 0; off >>= 1) { if (d < off) red[d] += red[d + off]; __syncthreads(); }
    const float rk = rsqrtf(red[0] * (1.f / 128.f) + EPS);

    const float qscale = 0.08838834764831845f;  // 1/sqrt(128), folded attention scale
    const size_t qi = ((size_t)bh * 2048 + s) * 128 + d;
    q[qi] = f2b(qv * rq * qscale);
    k[qi] = f2b(kv * rk);
    vT[((size_t)bh * 128 + d) * 2048 + s] = f2b(vv);
}

// ---------------------------------------------------------------------------
// Kernel 5: causal flash attention + fused (1/l)*rms_norm epilogue.
// Grid: B*H * (S/64). Block: 128 threads (4 waves), wave owns 16 q rows.
// Dynamic LDS layout (bytes):
//   Qs   [64][128] bf16 @ 0       (16384)
//   Ks   [64][128] bf16 @ 16384   (16384)
//   Vs   [128][64] bf16 @ 32768   (16384)   (Vs[d][kv] = V[kv][d])
//   Ss   4x[16][64] f32 @ 49152   (16384)
//   Ps   4x[16][64] bf16 @ 65536  (8192)
//   rowm f32[64] @ 73728, rowl @ 73984, rowa @ 74240  -> total 74496
//   Os   [64][128] f32 overlays Qs+Ks after the kv loop (32768)
// Tile staging uses async global->LDS copies (ASYNCcnt).
// ---------------------------------------------------------------------------
#define ATT_SMEM 74496

__global__ void __launch_bounds__(128) attn_kernel(const bf16_t* __restrict__ q,
                                                   const bf16_t* __restrict__ k,
                                                   const bf16_t* __restrict__ vT,
                                                   bf16_t* __restrict__ aout) {
    extern __shared__ char smem[];
    bf16_t* Qs   = (bf16_t*)(smem);
    bf16_t* Ks   = (bf16_t*)(smem + 16384);
    bf16_t* Vs   = (bf16_t*)(smem + 32768);
    float*  Ss   = (float*)(smem + 49152);
    bf16_t* Ps   = (bf16_t*)(smem + 65536);
    float*  rowm = (float*)(smem + 73728);
    float*  rowl = (float*)(smem + 73984);
    float*  rowa = (float*)(smem + 74240);
    float*  Os   = (float*)(smem);          // overlay, used after the loop

    const int tid  = threadIdx.x;
    const int lane = tid & 31;
    const int wave = tid >> 5;
    const int bh   = blockIdx.x >> 5;       // S/64 = 32 q-tiles
    const int qt   = blockIdx.x & 31;
    const int q0   = qt * 64;

    const bf16_t* qbase = q  + ((size_t)bh * 2048 + q0) * 128;
    const bf16_t* kbase = k  + (size_t)bh * 2048 * 128;
    const bf16_t* vbase = vT + (size_t)bh * 128 * 2048;

    for (int c = tid; c < 1024; c += 128)
        async_copy16(&qbase[c * 8], &Qs[c * 8]);
    if (tid < 64) { rowm[tid] = -1e30f; rowl[tid] = 0.f; }
    wait_async_0();
    __syncthreads();

    v8f oacc[8];
#pragma unroll
    for (int i = 0; i < 8; ++i) oacc[i] = zero8();

    const int myq = wave * 16;
    const int cr  = (lane < 16) ? 0 : 8;
    const int cc  = lane & 15;

    for (int kv0 = 0; kv0 < q0 + 64; kv0 += 64) {
        for (int c = tid; c < 1024; c += 128)
            async_copy16(&kbase[(size_t)kv0 * 128 + c * 8], &Ks[c * 8]);
        for (int c = tid; c < 1024; c += 128) {
            const int dr = c >> 3, ch = c & 7;
            async_copy16(&vbase[(size_t)dr * 2048 + kv0 + ch * 8], &Vs[dr * 64 + ch * 8]);
        }
        wait_async_0();
        __syncthreads();

        // S = Q (16x128) * K^T -> 16x64, K-dim in 4 steps of 32
        v8f sacc[4];
#pragma unroll
        for (int nt = 0; nt < 4; ++nt) sacc[nt] = zero8();
#pragma unroll
        for (int kk = 0; kk < 4; ++kk) {
            v16bf aq = frag_A(&Qs[myq * 128], 128, kk * 32, lane);
#pragma unroll
            for (int nt = 0; nt < 4; ++nt) {
                v16bf bk = frag_B(&Ks[nt * 16 * 128], 128, kk * 32, lane);
                sacc[nt] = wmma_bf16(aq, bk, sacc[nt]);
            }
        }
        float* Sw = Ss + wave * 16 * 64;
#pragma unroll
        for (int nt = 0; nt < 4; ++nt)
#pragma unroll
            for (int g = 0; g < 8; ++g)
                Sw[(cr + g) * 64 + nt * 16 + cc] = sacc[nt][g];
        __syncthreads();

        // online softmax, one thread per q row (threads 0..63)
        if (tid < 64) {
            const int r    = tid;
            const int grow = q0 + r;
            float* Sr  = Ss + (r >> 4) * (16 * 64) + (r & 15) * 64;
            bf16_t* Pr = Ps + (r >> 4) * (16 * 64) + (r & 15) * 64;
            const float mold = rowm[r];
            float tmax = -1e30f;
            for (int c2 = 0; c2 < 64; ++c2) {
                const float sv = (kv0 + c2 <= grow) ? Sr[c2] : -1e30f;
                tmax = fmaxf(tmax, sv);
            }
            const float mnew  = fmaxf(mold, tmax);
            const float alpha = __expf(mold - mnew);
            float lsum = 0.f;
            for (int c2 = 0; c2 < 64; ++c2) {
                const float p = (kv0 + c2 <= grow) ? __expf(Sr[c2] - mnew) : 0.f;
                lsum += p;
                Pr[c2] = f2b(p);
            }
            rowl[r] = rowl[r] * alpha + lsum;
            rowm[r] = mnew;
            rowa[r] = alpha;
        }
        __syncthreads();

        // rescale running output then accumulate P*V
        float af[8];
#pragma unroll
        for (int g = 0; g < 8; ++g) af[g] = rowa[myq + cr + g];
#pragma unroll
        for (int nt = 0; nt < 8; ++nt)
#pragma unroll
            for (int g = 0; g < 8; ++g) oacc[nt][g] *= af[g];

        bf16_t* Pw = Ps + wave * 16 * 64;
#pragma unroll
        for (int kk = 0; kk < 2; ++kk) {
            v16bf ap = frag_A(Pw, 64, kk * 32, lane);
#pragma unroll
            for (int nt = 0; nt < 8; ++nt) {
                v16bf bv = frag_B(&Vs[nt * 16 * 64], 64, kk * 32, lane);
                oacc[nt] = wmma_bf16(ap, bv, oacc[nt]);
            }
        }
        __syncthreads();
    }

    // stage O to LDS (overlays Qs/Ks, safe after loop)
#pragma unroll
    for (int nt = 0; nt < 8; ++nt)
#pragma unroll
        for (int g = 0; g < 8; ++g)
            Os[(myq + cr + g) * 128 + nt * 16 + cc] = oacc[nt][g];
    __syncthreads();

    // epilogue: y = O/l, rms_norm over hd, write bf16 to a[B,S,D]
    if (tid < 64) {
        const int r    = tid;
        const int grow = q0 + r;
        const float inv_l = 1.f / rowl[r];
        float ss = 0.f;
        for (int c2 = 0; c2 < 128; ++c2) {
            const float y = Os[r * 128 + c2] * inv_l;
            ss += y * y;
        }
        const float rn = rsqrtf(ss * (1.f / 128.f) + EPS) * inv_l;
        const int h = bh & 15, b = bh >> 4;
        bf16_t* outp = aout + ((size_t)(b * 2048 + grow)) * 2048 + h * 128;
        for (int c2 = 0; c2 < 128; ++c2) outp[c2] = f2b(Os[r * 128 + c2] * rn);
    }
}

// ---------------------------------------------------------------------------
// Host launch
// ---------------------------------------------------------------------------
extern "C" void kernel_launch(void* const* d_in, const int* in_sizes, int n_in,
                              void* d_out, int out_size, void* d_ws, size_t ws_size,
                              hipStream_t stream) {
    (void)in_sizes; (void)n_in; (void)out_size; (void)ws_size;
    const float* x     = (const float*)d_in[0];   // [4,2048,2048]
    const float* w_in  = (const float*)d_in[1];   // [6144,2048]
    const float* w_out = (const float*)d_in[2];   // [2048,2048]
    float* out = (float*)d_out;                   // [4,2048,2048]

    char* ws = (char*)d_ws;
    bf16_t* w_in_b  = (bf16_t*)(ws);                          // 6144*2048*2  = 25165824
    bf16_t* w_out_b = (bf16_t*)(ws + 25165824);               // 2048*2048*2  = 8388608
    bf16_t* x_b     = (bf16_t*)(ws + 33554432);               // 8192*2048*2  = 33554432
    bf16_t* qkv_b   = (bf16_t*)(ws + 67108864);               // 8192*6144*2  = 100663296
    bf16_t* q_b     = (bf16_t*)(ws + 167772160);              // 33554432
    bf16_t* k_b     = (bf16_t*)(ws + 201326592);              // 33554432
    bf16_t* vT_b    = (bf16_t*)(ws + 234881024);              // 33554432
    bf16_t* a_b     = (bf16_t*)(ws + 268435456);              // 33554432

    // 1) unit-norm weights -> bf16
    normw_kernel<<<6144, 256, 0, stream>>>(w_in,  w_in_b,  2048);
    normw_kernel<<<2048, 256, 0, stream>>>(w_out, w_out_b, 2048);
    // 2) x -> bf16
    cvt_kernel<<<(8192L * 2048L) / 256, 256, 0, stream>>>(x, x_b, 8192L * 2048L);
    // 3) qkv = x @ w_in^T  (M=8192, N=6144, K=2048), bf16 out
    gemm_wmma<true><<<64 * 48, 256, 0, stream>>>(x_b, w_in_b, nullptr, qkv_b, 8192, 6144, 2048);
    // 4) rms_norm q/k, build q,k,vT
    qkv_prep<<<4 * 16 * 2048, 128, 0, stream>>>(qkv_b, q_b, k_b, vT_b);
    // 5) causal flash attention + rms epilogue
    (void)hipFuncSetAttribute((const void*)attn_kernel,
                              hipFuncAttributeMaxDynamicSharedMemorySize, ATT_SMEM);
    attn_kernel<<<64 * 32, 128, ATT_SMEM, stream>>>(q_b, k_b, vT_b, a_b);
    // 6) out = a @ w_out^T  (M=8192, N=2048, K=2048), f32 out
    gemm_wmma<false><<<64 * 16, 256, 0, stream>>>(a_b, w_out_b, out, nullptr, 8192, 2048, 2048);
}
